// SelfMultiHeadAttention_73830487818494
// MI455X (gfx1250) — compile-verified, matrix operand order checked by reading, not run on hardware
//
#include <hip/hip_runtime.h>

typedef __attribute__((ext_vector_type(8)))  _Float16 v8h;
typedef __attribute__((ext_vector_type(16))) _Float16 v16h;
typedef __attribute__((ext_vector_type(8)))  float    v8f;

#define DMODEL 1024
#define NHEADS 16
#define DK     64
#define SEQ    2048
#define BATCH  4
#define MROWS  (BATCH * SEQ)   // 8192

// ---------------------------------------------------------------------------
// Fragment loader: 16x32 f16 tile (rows of length 32 along K, row stride ld),
// laid out per the CDNA5 ISA 16-bit A-matrix table:
//   lanes 0-15  : row = lane,    K = {0..7, 16..23}
//   lanes 16-31 : row = lane-16, K = {8..15, 24..31}
// Two 16-byte chunks per lane. B operand (N x K, K-contiguous) assumed
// symmetric with A (N plays the role of M), so the same loader serves both.
// ---------------------------------------------------------------------------
__device__ __forceinline__ v16h load_frag16x32(const _Float16* base, int ld) {
  const int lane = threadIdx.x & 31;
  const _Float16* p = base + (lane & 15) * ld + ((lane >> 4) << 3);
  v8h lo = *(const v8h*)(p);
  v8h hi = *(const v8h*)(p + 16);
  v16h a;
#pragma unroll
  for (int i = 0; i < 8; ++i) { a[i] = lo[i]; a[i + 8] = hi[i]; }
  return a;
}

__device__ __forceinline__ v8f wmma32(v16h a, v16h b, v8f c) {
  return __builtin_amdgcn_wmma_f32_16x16x32_f16(
      /*neg_a=*/false, a, /*neg_b=*/false, b,
      /*c_mod=*/(short)0, c, /*reuse_a=*/false, /*reuse_b=*/false);
}

// Async global -> LDS copy of 16 bytes (CDNA5 GLOBAL_LOAD_ASYNC_TO_LDS_B128,
// tracked by ASYNCcnt; no VGPR round-trip). lds_off = LDS byte offset
// (= low 32 bits of the generic shared-aperture address).
__device__ __forceinline__ void async_copy_b128(void* lds_ptr, const void* gptr) {
  unsigned lds_off = (unsigned)(size_t)lds_ptr;
  asm volatile("global_load_async_to_lds_b128 %0, %1, off"
               :: "v"(lds_off), "v"(gptr) : "memory");
}
template <int N>
__device__ __forceinline__ void wait_asynccnt() {
  asm volatile("s_wait_asynccnt %0" :: "i"(N) : "memory");
}

// ---------------------------------------------------------------------------
// fp32 -> f16 conversion (grid-stride)
// ---------------------------------------------------------------------------
__global__ void cvt_f32_f16(const float* __restrict__ src,
                            _Float16* __restrict__ dst, int n) {
  int i = blockIdx.x * blockDim.x + threadIdx.x;
  int stride = gridDim.x * blockDim.x;
  for (; i < n; i += stride) dst[i] = (_Float16)src[i];
}

// ---------------------------------------------------------------------------
// GEMM: C[m][n] = sum_k A[m][k] * W[n][k] + bias[n]
//   A: [8192][1024] f16 row-major, W: [1024][1024] f16 row-major
// mode 0: store f16 head-split  out[((b*16+h)*2048+s)*64 + d]      (Q, K)
// mode 1: store f16 head-split transposed out[((b*16+h)*64+d)*2048 + s] (V^T)
// mode 2: store f32 plain out[m*1024+n]                            (final)
// One wave computes a 32x64 tile: 2 A-frags x 4 B-frags = 8 WMMAs per k-step.
// ---------------------------------------------------------------------------
__global__ void gemm_f16_kernel(const _Float16* __restrict__ A,
                                const _Float16* __restrict__ W,
                                const float* __restrict__ bias,
                                void* __restrict__ out, int mode) {
  const int lane = threadIdx.x & 31;
  const int wave = threadIdx.x >> 5;
  const int wid  = blockIdx.x * 8 + wave;         // 0..4095
  const int mt   = (wid >> 4) << 5;               // row tile base (32 rows)
  const int nt   = (wid & 15) << 6;               // col tile base (64 cols)

  v8f c[2][4];
#pragma unroll
  for (int rf = 0; rf < 2; ++rf)
#pragma unroll
    for (int nf = 0; nf < 4; ++nf) c[rf][nf] = {};

  for (int k = 0; k < DMODEL; k += 32) {
    v16h a0 = load_frag16x32(A + (size_t)mt * DMODEL + k, DMODEL);
    v16h a1 = load_frag16x32(A + (size_t)(mt + 16) * DMODEL + k, DMODEL);
#pragma unroll
    for (int nf = 0; nf < 4; ++nf) {
      v16h b = load_frag16x32(W + (size_t)(nt + nf * 16) * DMODEL + k, DMODEL);
      c[0][nf] = wmma32(a0, b, c[0][nf]);
      c[1][nf] = wmma32(a1, b, c[1][nf]);
    }
  }

  // C layout: element i of frag -> row = i + 8*(lane>=16), col = lane&15
  const int rb = (lane >> 4) << 3;
  const int nl = lane & 15;
#pragma unroll
  for (int nf = 0; nf < 4; ++nf) {
    const int n  = nt + nf * 16 + nl;
    const float bv = bias[n];
#pragma unroll
    for (int rf = 0; rf < 2; ++rf) {
#pragma unroll
      for (int i = 0; i < 8; ++i) {
        const int m = mt + rf * 16 + rb + i;
        const float val = c[rf][nf][i] + bv;
        if (mode == 2) {
          ((float*)out)[(size_t)m * DMODEL + n] = val;
        } else {
          const int b = m >> 11, s = m & (SEQ - 1);
          const int h = n >> 6,  d = n & (DK - 1);
          const _Float16 hv = (_Float16)val;
          if (mode == 0)
            ((_Float16*)out)[(((size_t)b * NHEADS + h) * SEQ + s) * DK + d] = hv;
          else
            ((_Float16*)out)[(((size_t)b * NHEADS + h) * DK + d) * SEQ + s] = hv;
        }
      }
    }
  }
}

// ---------------------------------------------------------------------------
// Flash attention. Grid: (B*H, SEQ/128). Block: 256 threads = 8 waves.
// Each wave: one 16-row query tile, streams 32-key blocks.
//   Q  : [bh][s][64]  f16
//   K  : [bh][s][64]  f16
//   Vt : [bh][d][2048] f16 (transposed V)
//   O  : [b][s][h*64+d] f16 (row-major 8192x1024, feeds final GEMM)
// K/Vt tiles double-buffered in LDS via async global->LDS copies: tile j+1
// streams in (ASYNCcnt in flight) while tile j is consumed by the WMMAs.
// ---------------------------------------------------------------------------
__global__ void attn_kernel(const _Float16* __restrict__ Q,
                            const _Float16* __restrict__ K,
                            const _Float16* __restrict__ Vt,
                            _Float16* __restrict__ O) {
  __shared__ __align__(16) _Float16 ldsK[2][32 * 64];   // [buf][key][d]
  __shared__ __align__(16) _Float16 ldsVt[2][64 * 32];  // [buf][d][key]
  __shared__ __align__(16) _Float16 ldsP[8][16 * 32];   // per-wave P tile

  const int tid  = threadIdx.x;
  const int lane = tid & 31;
  const int wave = tid >> 5;
  const int bh   = blockIdx.x;                       // 0..63
  const int q0   = blockIdx.y * 128 + wave * 16;

  const _Float16* Qb = Q + ((size_t)bh * SEQ + q0) * DK;
  const v16h qf0 = load_frag16x32(Qb, DK);
  const v16h qf1 = load_frag16x32(Qb + 32, DK);

  v8f o[4];
#pragma unroll
  for (int nf = 0; nf < 4; ++nf) o[nf] = {};
  float mrow[8], lrow[8];
#pragma unroll
  for (int i = 0; i < 8; ++i) { mrow[i] = -3.0e38f; lrow[i] = 0.0f; }

  const _Float16* Kb  = K + (size_t)bh * SEQ * DK;
  const _Float16* Vtb = Vt + (size_t)bh * DK * SEQ;

  const int ksl = tid >> 3;            // 0..31  (key row for K staging)
  const int kc8 = (tid & 7) << 3;      // 0..56  (d chunk)
  const int vdl = tid >> 2;            // 0..63  (d row for Vt staging)
  const int vc8 = (tid & 3) << 3;      // 0..24  (key chunk)
  const int rb  = (lane >> 4) << 3;
  const int nl  = lane & 15;

  // per-thread async staging of one key-block tile (2 x b128 per thread)
  auto stage = [&](int tile, int buf) {
    async_copy_b128(&ldsK[buf][ksl * 64 + kc8],
                    Kb + (size_t)(tile * 32 + ksl) * DK + kc8);
    async_copy_b128(&ldsVt[buf][vdl * 32 + vc8],
                    Vtb + (size_t)vdl * SEQ + tile * 32 + vc8);
  };

  const int NT = SEQ / 32;             // 64 key-block tiles
  stage(0, 0);                         // prologue: tile 0 -> buffer 0

  for (int it = 0; it < NT; ++it) {
    const int cur = it & 1;
    if (it + 1 < NT) {
      stage(it + 1, cur ^ 1);          // prefetch next tile into other buffer
      wait_asynccnt<2>();              // oldest 2 (current tile) have landed
    } else {
      wait_asynccnt<0>();
    }
    __syncthreads();                   // all waves' current-tile copies visible

    // ---- scores: 16 queries x 32 keys, f32 accum ----
    const _Float16* Kc  = &ldsK[cur][0];
    const _Float16* Vtc = &ldsVt[cur][0];
    v8f s0 = {}, s1 = {};
    s0 = wmma32(qf0, load_frag16x32(Kc, 64), s0);
    s0 = wmma32(qf1, load_frag16x32(Kc + 32, 64), s0);
    s1 = wmma32(qf0, load_frag16x32(Kc + 16 * 64, 64), s1);
    s1 = wmma32(qf1, load_frag16x32(Kc + 16 * 64 + 32, 64), s1);

    // ---- online softmax (row = frag element i within a lane half) ----
#pragma unroll
    for (int i = 0; i < 8; ++i) {
      float v0 = s0[i] * 0.125f;   // 1/sqrt(64)
      float v1 = s1[i] * 0.125f;
      float mx = fmaxf(v0, v1);
#pragma unroll
      for (int off = 8; off; off >>= 1) mx = fmaxf(mx, __shfl_xor(mx, off, 32));
      const float mn   = fmaxf(mrow[i], mx);
      const float corr = __expf(mrow[i] - mn);
      const float p0   = __expf(v0 - mn);
      const float p1   = __expf(v1 - mn);
      s0[i] = p0; s1[i] = p1;
      float rs = p0 + p1;
#pragma unroll
      for (int off = 8; off; off >>= 1) rs += __shfl_xor(rs, off, 32);
      lrow[i] = lrow[i] * corr + rs;
      mrow[i] = mn;
      o[0][i] *= corr; o[1][i] *= corr; o[2][i] *= corr; o[3][i] *= corr;
    }

    // ---- reshape P (C layout) -> A layout via per-wave LDS tile ----
    _Float16* Pw = &ldsP[wave][0];
#pragma unroll
    for (int i = 0; i < 8; ++i) {
      Pw[(rb + i) * 32 + nl]      = (_Float16)s0[i];
      Pw[(rb + i) * 32 + 16 + nl] = (_Float16)s1[i];
    }
    __syncthreads();

    // ---- O += P (16x32) @ V (32x64), B from transposed V in LDS ----
    const v16h pf = load_frag16x32(Pw, 32);
#pragma unroll
    for (int nf = 0; nf < 4; ++nf)
      o[nf] = wmma32(pf, load_frag16x32(Vtc + nf * 16 * 32, 32), o[nf]);
    __syncthreads();   // all waves done reading buf[cur] before it is re-staged
  }

  // ---- epilogue: normalize and store f16 row-major [b][s][h*64+d] ----
  const int b = bh >> 4, h = bh & (NHEADS - 1);
#pragma unroll
  for (int nf = 0; nf < 4; ++nf) {
#pragma unroll
    for (int i = 0; i < 8; ++i) {
      const int q = q0 + rb + i;
      const float val = o[nf][i] / lrow[i];
      O[((size_t)b * SEQ + q) * DMODEL + h * DK + nf * 16 + nl] = (_Float16)val;
    }
  }
}

// ---------------------------------------------------------------------------
// Host launcher
// ---------------------------------------------------------------------------
extern "C" void kernel_launch(void* const* d_in, const int* in_sizes, int n_in,
                              void* d_out, int out_size, void* d_ws, size_t ws_size,
                              hipStream_t stream) {
  (void)in_sizes; (void)n_in; (void)out_size; (void)ws_size;
  const float* x  = (const float*)d_in[0];
  const float* Wq = (const float*)d_in[1];
  const float* bq = (const float*)d_in[2];
  const float* Wk = (const float*)d_in[3];
  const float* bk = (const float*)d_in[4];
  const float* Wv = (const float*)d_in[5];
  const float* bv = (const float*)d_in[6];
  const float* Wo = (const float*)d_in[7];
  const float* bo = (const float*)d_in[8];

  _Float16* w = (_Float16*)d_ws;
  size_t off = 0;
  _Float16* xh  = w + off; off += (size_t)MROWS * DMODEL;        // 8M halfs
  _Float16* Wqh = w + off; off += (size_t)DMODEL * DMODEL;       // 1M
  _Float16* Wkh = w + off; off += (size_t)DMODEL * DMODEL;
  _Float16* Wvh = w + off; off += (size_t)DMODEL * DMODEL;
  _Float16* Woh = w + off; off += (size_t)DMODEL * DMODEL;
  _Float16* Qh  = w + off; off += (size_t)BATCH * NHEADS * SEQ * DK;
  _Float16* Kh  = w + off; off += (size_t)BATCH * NHEADS * SEQ * DK;
  _Float16* Vth = w + off; off += (size_t)BATCH * NHEADS * SEQ * DK;
  _Float16* AOh = w + off; off += (size_t)MROWS * DMODEL;

  // 1) fp32 -> f16 conversions
  cvt_f32_f16<<<2048, 256, 0, stream>>>(x,  xh,  MROWS * DMODEL);
  cvt_f32_f16<<<512,  256, 0, stream>>>(Wq, Wqh, DMODEL * DMODEL);
  cvt_f32_f16<<<512,  256, 0, stream>>>(Wk, Wkh, DMODEL * DMODEL);
  cvt_f32_f16<<<512,  256, 0, stream>>>(Wv, Wvh, DMODEL * DMODEL);
  cvt_f32_f16<<<512,  256, 0, stream>>>(Wo, Woh, DMODEL * DMODEL);

  // 2) QKV projections (WMMA GEMMs); V stored transposed per head
  gemm_f16_kernel<<<512, 256, 0, stream>>>(xh, Wqh, bq, (void*)Qh, 0);
  gemm_f16_kernel<<<512, 256, 0, stream>>>(xh, Wkh, bk, (void*)Kh, 0);
  gemm_f16_kernel<<<512, 256, 0, stream>>>(xh, Wvh, bv, (void*)Vth, 1);

  // 3) flash attention (WMMA + double-buffered async LDS staging)
  attn_kernel<<<dim3(BATCH * NHEADS, SEQ / 128), 256, 0, stream>>>(Qh, Kh, Vth, AOh);

  // 4) output projection, fp32 result + bias into d_out
  gemm_f16_kernel<<<512, 256, 0, stream>>>(AOh, Woh, bo, d_out, 2);
}
